// LoraLinear_15015205667342
// MI455X (gfx1250) — matrix operand order repaired and checked by reading, not compiled, not run on hardware
//
#include <hip/hip_runtime.h>

typedef __attribute__((ext_vector_type(16))) __bf16 v16bf;
typedef __attribute__((ext_vector_type(8)))  __bf16 v8bf;
typedef __attribute__((ext_vector_type(8)))  float  v8f;

#define TOK    8192
#define DIM    4096
#define OUTD   4096
#define RRANK  64
#define SCALING 2.0f

// bitsandbytes NF4 codebook
__constant__ float NF4C[16] = {
    -1.0f, -0.6961928009986877f, -0.5250730514526367f, -0.39491748809814453f,
    -0.28444138169288635f, -0.18477343022823334f, -0.09105003625154495f, 0.0f,
    0.07958029955625534f, 0.16093020141124725f, 0.24611230194568634f, 0.33791524171829224f,
    0.44070982933044434f, 0.5626170039176941f, 0.7229568362236023f, 1.0f };

// ---------------------------------------------------------------------------
// Kernel 1: XA[t,r] = SCALING * sum_d x[t,d] * lora_A[r,d]   (bf16 out, [8192,64])
// One 16x16 tile per wave, K=4096 via v_wmma_f32_16x16x32_bf16.
// ---------------------------------------------------------------------------
__global__ __launch_bounds__(256) void xa_gemm_kernel(
    const float* __restrict__ x, const float* __restrict__ loraA,
    unsigned short* __restrict__ xa_out)
{
  __bf16* xa = (__bf16*)xa_out;
  const int tid  = threadIdx.x;
  const int lane = tid & 31;
  const int wave = tid >> 5;
  const int tile = blockIdx.x * 8 + wave;   // 2048 tiles total
  const int tm   = tile >> 2;               // token tile: 0..511
  const int tn   = tile & 3;                // r tile:     0..3
  const int m    = lane & 15;
  const int half = lane >> 4;

  const float* xrow = x     + (size_t)(tm * 16 + m) * DIM;
  const float* arow = loraA + (size_t)(tn * 16 + m) * DIM;

  v8f acc = {};
  for (int kk = 0; kk < DIM; kk += 32) {
    const int ka = kk + half * 16;    // A: 16 contiguous K per half-wave
    const int kb = kk + half * 16;    // B: 16 contiguous K per half-wave
    v16bf a, b;
#pragma unroll
    for (int e = 0; e < 8; ++e) {
      a[e]     = (__bf16)xrow[ka + e];
      a[e + 8] = (__bf16)xrow[ka + 8 + e];
    }
#pragma unroll
    for (int e = 0; e < 16; ++e) b[e] = (__bf16)arow[kb + e];
    acc = __builtin_amdgcn_wmma_f32_16x16x32_bf16(false, a, false, b,
                                                  (short)0, acc, false, false);
  }
  const int rowoff = half * 8;
#pragma unroll
  for (int v = 0; v < 8; ++v) {
    const int t = tm * 16 + rowoff + v;
    const int r = tn * 16 + m;
    xa[(size_t)t * RRANK + r] = (__bf16)(SCALING * acc[v]);
  }
}

// ---------------------------------------------------------------------------
// Kernel 2: out = x * dequant_nf4(W)^T + XA * lora_B^T
// Block tile 128 tok x 64 out, 8 waves (each 32x32 = 2x2 WMMA accumulators).
// Per K-tile (BK=64): stage x->bf16 and dequantized W->bf16 in LDS.
// absmax block size (64) == BK, so one scale per W row per K-tile.
// ---------------------------------------------------------------------------
#define BM 128
#define BN 64
#define BK 64
#define XSTR 72   // padded bf16 row stride (144B, 16B aligned)
#define WSTR 72

__global__ __launch_bounds__(256) void lora_nf4_main_kernel(
    const float* __restrict__ x, const int* __restrict__ codes,
    const float* __restrict__ absmax, const float* __restrict__ loraB,
    const unsigned short* __restrict__ xa_in, float* __restrict__ out)
{
  const __bf16* xa = (const __bf16*)xa_in;
  __shared__ __align__(16) __bf16 xs[BM * XSTR];
  __shared__ __align__(16) __bf16 wsh[BN * WSTR];
  __shared__ float nf4s[16];

  const int tid  = threadIdx.x;
  const int lane = tid & 31;
  const int wave = tid >> 5;
  const int wr   = wave & 3;    // token quadrant (32 tokens)
  const int wc   = wave >> 2;   // out half (32 outs)
  const int m    = lane & 15;
  const int half = lane >> 4;

  const int tokBase = blockIdx.x * BM;
  const int outBase = blockIdx.y * BN;

  if (tid < 16) nf4s[tid] = NF4C[tid];

  v8f acc[2][2] = {};

  for (int kk = 0; kk < DIM; kk += BK) {
    __syncthreads();
    // ---- stage x tile: 128 x 64 f32 -> bf16 (8 float4 per thread) ----
#pragma unroll
    for (int i = 0; i < 8; ++i) {
      const int idx = tid + i * 256;
      const int r   = idx >> 4;
      const int c4  = (idx & 15) * 4;
      const float4 v = *(const float4*)(x + (size_t)(tokBase + r) * DIM + kk + c4);
      __bf16* dst = &xs[r * XSTR + c4];
      dst[0] = (__bf16)v.x; dst[1] = (__bf16)v.y;
      dst[2] = (__bf16)v.z; dst[3] = (__bf16)v.w;
    }
    // ---- stage + dequant W tile: 64 x 64 codes -> bf16 (4 int4 per thread) ----
#pragma unroll
    for (int i = 0; i < 4; ++i) {
      const int idx = tid + i * 256;
      const int r   = idx >> 4;
      const int c4  = (idx & 15) * 4;
      const int o   = outBase + r;
      const int4 cv = *(const int4*)(codes + (size_t)o * DIM + kk + c4);
      const float s = absmax[o * (DIM / 64) + (kk >> 6)];
      __bf16* dst = &wsh[r * WSTR + c4];
      dst[0] = (__bf16)(nf4s[cv.x] * s); dst[1] = (__bf16)(nf4s[cv.y] * s);
      dst[2] = (__bf16)(nf4s[cv.z] * s); dst[3] = (__bf16)(nf4s[cv.w] * s);
    }
    __syncthreads();
    // ---- 2 WMMA K-steps over the staged 64-wide tile ----
#pragma unroll
    for (int ks = 0; ks < BK; ks += 32) {
      v16bf afr[2], bfr[2];
#pragma unroll
      for (int i = 0; i < 2; ++i) {
        const __bf16* p = &xs[(wr * 32 + i * 16 + m) * XSTR + ks + half * 16];
        v8bf lo = *(const v8bf*)p;
        v8bf hi = *(const v8bf*)(p + 8);
        afr[i] = __builtin_shufflevector(lo, hi, 0,1,2,3,4,5,6,7,8,9,10,11,12,13,14,15);
      }
#pragma unroll
      for (int j = 0; j < 2; ++j) {
        const __bf16* p = &wsh[(wc * 32 + j * 16 + m) * WSTR + ks + half * 16];
        v8bf lo = *(const v8bf*)p;
        v8bf hi = *(const v8bf*)(p + 8);
        bfr[j] = __builtin_shufflevector(lo, hi, 0,1,2,3,4,5,6,7,8,9,10,11,12,13,14,15);
      }
#pragma unroll
      for (int i = 0; i < 2; ++i)
#pragma unroll
        for (int j = 0; j < 2; ++j)
          acc[i][j] = __builtin_amdgcn_wmma_f32_16x16x32_bf16(
              false, afr[i], false, bfr[j], (short)0, acc[i][j], false, false);
    }
  }

  // ---- LoRA epilogue: acc += XA(16 x 64) * lora_B^T (K = 64, 2 WMMA steps) ----
#pragma unroll
  for (int ks = 0; ks < RRANK; ks += 32) {
    v16bf afr[2], bfr[2];
#pragma unroll
    for (int i = 0; i < 2; ++i) {
      const __bf16* p = xa + (size_t)(tokBase + wr * 32 + i * 16 + m) * RRANK + ks + half * 16;
      v8bf lo = *(const v8bf*)p;
      v8bf hi = *(const v8bf*)(p + 8);
      afr[i] = __builtin_shufflevector(lo, hi, 0,1,2,3,4,5,6,7,8,9,10,11,12,13,14,15);
    }
#pragma unroll
    for (int j = 0; j < 2; ++j) {
      const float* p = loraB + (size_t)(outBase + wc * 32 + j * 16 + m) * RRANK + ks + half * 16;
      v16bf b;
#pragma unroll
      for (int e = 0; e < 16; ++e) b[e] = (__bf16)p[e];
      bfr[j] = b;
    }
#pragma unroll
    for (int i = 0; i < 2; ++i)
#pragma unroll
      for (int j = 0; j < 2; ++j)
        acc[i][j] = __builtin_amdgcn_wmma_f32_16x16x32_bf16(
            false, afr[i], false, bfr[j], (short)0, acc[i][j], false, false);
  }

  // ---- store f32 output per the 16x16 f32 C/D layout ----
  const int rowoff = half * 8;
#pragma unroll
  for (int i = 0; i < 2; ++i)
#pragma unroll
    for (int j = 0; j < 2; ++j)
#pragma unroll
      for (int v = 0; v < 8; ++v) {
        const int t = tokBase + wr * 32 + i * 16 + rowoff + v;
        const int o = outBase + wc * 32 + j * 16 + m;
        out[(size_t)t * OUTD + o] = acc[i][j][v];
      }
}

extern "C" void kernel_launch(void* const* d_in, const int* in_sizes, int n_in,
                              void* d_out, int out_size, void* d_ws, size_t ws_size,
                              hipStream_t stream) {
  const float* x      = (const float*)d_in[0];
  const float* lora_A = (const float*)d_in[1];
  const float* lora_B = (const float*)d_in[2];
  const float* absmax = (const float*)d_in[3];
  const int*   codes  = (const int*)d_in[4];
  float*       out    = (float*)d_out;
  unsigned short* xa  = (unsigned short*)d_ws;   // bf16 [8192, 64] = 1 MB

  xa_gemm_kernel<<<dim3(256), dim3(256), 0, stream>>>(x, lora_A, xa);
  lora_nf4_main_kernel<<<dim3(TOK / BM, OUTD / BN), dim3(256), 0, stream>>>(
      x, codes, absmax, lora_B, xa, out);
}
